// Attention_51891794870886
// MI455X (gfx1250) — matrix operand order repaired and checked by reading, not compiled, not run on hardware
//
#include <hip/hip_runtime.h>
#include <hip/hip_bf16.h>

typedef __bf16 bf16;
typedef __attribute__((ext_vector_type(16))) __bf16 v16bf;
typedef __attribute__((ext_vector_type(8)))  __bf16 v8bf;   // 16-byte chunk
typedef __attribute__((ext_vector_type(8)))  float  v8f;

union FragAB { v16bf v; v8bf h[2]; };

__device__ __forceinline__ bf16 f2bf(float f) {
    unsigned u = __float_as_uint(f);
    unsigned r = (u + 0x7FFFu + ((u >> 16) & 1u)) >> 16;   // round-to-nearest-even
    unsigned short s = (unsigned short)r;
    return __builtin_bit_cast(bf16, s);
}

__device__ __forceinline__ v8f wmma_bf16(v16bf a, v16bf b, v8f c) {
    return __builtin_amdgcn_wmma_f32_16x16x32_bf16(
        /*neg_a=*/false, a, /*neg_b=*/false, b,
        /*c_mod=*/(short)0, c, /*reuse_a=*/false, /*reuse_b=*/false);
}

// CDNA5 async memory->LDS copy (ASYNCcnt-tracked, no VGPR transit).
// LDS dest = low 32 bits of generic pointer (flat->LDS aperture truncation).
__device__ __forceinline__ void async_copy_b128(void* lds_ptr, const void* gptr) {
    unsigned lds_off = (unsigned)(unsigned long long)lds_ptr;
    unsigned long long ga = (unsigned long long)gptr;
    asm volatile("global_load_async_to_lds_b128 %0, %1, off"
                 :: "v"(lds_off), "v"(ga) : "memory");
}
__device__ __forceinline__ void wait_async0() {
    asm volatile("s_wait_asynccnt 0" ::: "memory");
}

__device__ __forceinline__ float rsum32(float v) {
#pragma unroll
    for (int m = 16; m >= 1; m >>= 1) v += __shfl_xor(v, m, 32);
    return v;
}
__device__ __forceinline__ float rmax16(float v) {
#pragma unroll
    for (int m = 8; m >= 1; m >>= 1) v = fmaxf(v, __shfl_xor(v, m, 16));
    return v;
}
__device__ __forceinline__ float rsum16(float v) {
#pragma unroll
    for (int m = 8; m >= 1; m >>= 1) v += __shfl_xor(v, m, 16);
    return v;
}

// ---------------------------------------------------------------------------
// fp32 -> bf16 elementwise conversion (grid-stride)
// ---------------------------------------------------------------------------
__global__ void f32_to_bf16_kernel(const float* __restrict__ in,
                                   bf16* __restrict__ out, int n) {
    for (int i = blockIdx.x * blockDim.x + threadIdx.x; i < n;
         i += gridDim.x * blockDim.x)
        out[i] = f2bf(in[i]);
}

// ---------------------------------------------------------------------------
// C[M,N] (f32, +optional bias) = A[M,K] (bf16 row-major) * B[N,K]^T (bf16 row-major)
// Block tile 64x128, 8 waves, each wave 32x32 (2x2 WMMA tiles), K step 32.
// Double-buffered LDS staged with global_load_async_to_lds_b128.
// ---------------------------------------------------------------------------
__global__ __launch_bounds__(256)
void gemm_bf16_kernel(const bf16* __restrict__ A, const bf16* __restrict__ B,
                      const float* __restrict__ bias, float* __restrict__ C,
                      int M, int N, int K) {
    __shared__ __align__(16) bf16 a_lds[2][64 * 32];
    __shared__ __align__(16) bf16 b_lds[2][128 * 32];

    const int tid  = threadIdx.x;
    const int lane = tid & 31, w = tid >> 5;
    const int wr = w >> 2, wc = w & 3;            // wave grid 2x4
    const int m0 = blockIdx.x * 64, n0 = blockIdx.y * 128;
    const int arow = lane & 15, half = lane >> 4;

    v8f acc[2][2] = {};

    auto issue_tile = [&](int buf, int k0) {
        {   // A tile: 64x32 -> 256 chunks of 8 bf16
            int r = tid >> 2, c = (tid & 3) * 8;
            async_copy_b128(&a_lds[buf][r * 32 + c],
                            &A[(size_t)(m0 + r) * K + k0 + c]);
        }
#pragma unroll
        for (int i = 0; i < 2; ++i) {  // B tile: 128x32 -> 512 chunks
            int ch = tid + i * 256;
            int r = ch >> 2, c = (ch & 3) * 8;
            async_copy_b128(&b_lds[buf][r * 32 + c],
                            &B[(size_t)(n0 + r) * K + k0 + c]);
        }
    };

    issue_tile(0, 0);
    wait_async0();
    __syncthreads();

    const int nk = K >> 5;
    for (int kt = 0; kt < nk; ++kt) {
        if (kt + 1 < nk) issue_tile((kt + 1) & 1, (kt + 1) * 32);

        const bf16* al = a_lds[kt & 1];
        const bf16* bl = b_lds[kt & 1];
        FragAB af[2], bfr[2];
#pragma unroll
        for (int i = 0; i < 2; ++i) {  // A frag: lane<16 K{0..7,16..23}, lane>=16 K{8..15,24..31}
            int row = wr * 32 + i * 16 + arow;
            af[i].h[0] = *(const v8bf*)&al[row * 32 + half * 8];
            af[i].h[1] = *(const v8bf*)&al[row * 32 + half * 8 + 16];
        }
#pragma unroll
        for (int j = 0; j < 2; ++j) {  // B frag: lane<16 K{0..7,16..23}, lane>=16 K{8..15,24..31}
            int col = wc * 32 + j * 16 + arow;
            bfr[j].h[0] = *(const v8bf*)&bl[col * 32 + half * 8];
            bfr[j].h[1] = *(const v8bf*)&bl[col * 32 + half * 8 + 16];
        }
#pragma unroll
        for (int i = 0; i < 2; ++i)
#pragma unroll
            for (int j = 0; j < 2; ++j)
                acc[i][j] = wmma_bf16(af[i].v, bfr[j].v, acc[i][j]);

        wait_async0();        // next tile resident
        __syncthreads();      // all waves done reading current buffer
    }

#pragma unroll
    for (int i = 0; i < 2; ++i)
#pragma unroll
        for (int j = 0; j < 2; ++j) {
            int rb = m0 + wr * 32 + i * 16 + half * 8;  // C layout: vgpr r -> row r (+8 hi half)
            int cb = n0 + wc * 32 + j * 16 + arow;
            float bv = bias ? bias[cb] : 0.f;
#pragma unroll
            for (int r = 0; r < 8; ++r)
                C[(size_t)(rb + r) * N + cb] = acc[i][j][r] + bv;
        }
}

// ---------------------------------------------------------------------------
// Per-(b,h,n) LayerNorm over D=64 for q,k; scale q by 1/8; pack bf16.
// q,k -> [bh][n][d]; v -> transposed [bh][d][n]. One wave per row.
// ---------------------------------------------------------------------------
__global__ __launch_bounds__(256)
void ln_pack_kernel(const float* __restrict__ qkv,
                    const float* __restrict__ qg, const float* __restrict__ qb,
                    const float* __restrict__ kg, const float* __restrict__ kb,
                    bf16* __restrict__ qo, bf16* __restrict__ ko,
                    bf16* __restrict__ vto) {
    const int w = threadIdx.x >> 5, lane = threadIdx.x & 31;
    const int R  = blockIdx.x * 8 + w;           // (b*H + h)*N + n
    const int n  = R & 2047, bh = R >> 11;
    const int b  = bh >> 4,  h  = bh & 15;
    const float* base = qkv + (size_t)(b * 2048 + n) * 3072 + h * 64;
    const int d0 = lane, d1 = lane + 32;
    const size_t ob = ((size_t)bh * 2048 + n) * 64;

    {   // Q: LN + scale 1/sqrt(D)
        float x0 = base[d0], x1 = base[d1];
        float mu = rsum32(x0 + x1) * (1.f / 64.f);
        float e0 = x0 - mu, e1 = x1 - mu;
        float var = rsum32(e0 * e0 + e1 * e1) * (1.f / 64.f);
        float inv = rsqrtf(var + 1e-6f);
        qo[ob + d0] = f2bf((e0 * inv * qg[d0] + qb[d0]) * 0.125f);
        qo[ob + d1] = f2bf((e1 * inv * qg[d1] + qb[d1]) * 0.125f);
    }
    {   // K: LN
        float x0 = base[1024 + d0], x1 = base[1024 + d1];
        float mu = rsum32(x0 + x1) * (1.f / 64.f);
        float e0 = x0 - mu, e1 = x1 - mu;
        float var = rsum32(e0 * e0 + e1 * e1) * (1.f / 64.f);
        float inv = rsqrtf(var + 1e-6f);
        ko[ob + d0] = f2bf(e0 * inv * kg[d0] + kb[d0]);
        ko[ob + d1] = f2bf(e1 * inv * kg[d1] + kb[d1]);
    }
    {   // V: pre-transposed store for the PV GEMM
        vto[((size_t)bh * 64 + d0) * 2048 + n] = f2bf(base[2048 + d0]);
        vto[((size_t)bh * 64 + d1) * 2048 + n] = f2bf(base[2048 + d1]);
    }
}

// ---------------------------------------------------------------------------
// Flash attention per (b,h): N=2048, D=64. Block = 64 query rows, 4 waves,
// each wave owns 16 rows. Online softmax, bf16 WMMA, fp32 accumulators.
// K/Vt tiles double-buffered via async-to-LDS copies.
// ---------------------------------------------------------------------------
__global__ __launch_bounds__(128)
void flash_attn_kernel(const bf16* __restrict__ Q, const bf16* __restrict__ Kr,
                       const bf16* __restrict__ Vt, bf16* __restrict__ O) {
    __shared__ __align__(16) bf16 k_lds[2][64 * 64];     // [key][d]
    __shared__ __align__(16) bf16 vt_lds[2][64 * 64];    // [d][key]
    __shared__ __align__(16) bf16 p_lds[4][16 * 64];     // per-wave P scratch

    const int tid = threadIdx.x, lane = tid & 31, w = tid >> 5;
    const int bh = blockIdx.y;
    const int q0 = blockIdx.x * 64 + w * 16;
    const int arow = lane & 15, half = lane >> 4;

    const bf16* Kb = Kr + (size_t)bh * 2048 * 64;
    const bf16* Vb = Vt + (size_t)bh * 64 * 2048;

    // Q fragments (whole D=64 = two 16x32 A-frags), resident in registers.
    const size_t qrow = (size_t)bh * 2048 + q0 + arow;
    FragAB qa[2];
#pragma unroll
    for (int c = 0; c < 2; ++c) {
        qa[c].h[0] = *(const v8bf*)&Q[qrow * 64 + c * 32 + half * 8];
        qa[c].h[1] = *(const v8bf*)&Q[qrow * 64 + c * 32 + half * 8 + 16];
    }

    float mrow[8], lrow[8];
    v8f oacc[4] = {};
#pragma unroll
    for (int r = 0; r < 8; ++r) { mrow[r] = -3.0e38f; lrow[r] = 0.f; }

    auto issue_kv = [&](int buf, int kb0) {
#pragma unroll
        for (int i = 0; i < 4; ++i) {   // K tile 64x64 keys x d
            int ch = tid + i * 128;
            int r = ch >> 3, c = (ch & 7) * 8;
            async_copy_b128(&k_lds[buf][r * 64 + c],
                            &Kb[(size_t)(kb0 + r) * 64 + c]);
        }
#pragma unroll
        for (int i = 0; i < 4; ++i) {   // Vt tile 64(d) x 64(key)
            int ch = tid + i * 128;
            int r = ch >> 3, c = (ch & 7) * 8;
            async_copy_b128(&vt_lds[buf][r * 64 + c],
                            &Vb[(size_t)r * 2048 + kb0 + c]);
        }
    };

    issue_kv(0, 0);
    wait_async0();
    __syncthreads();

    const int nkb = 2048 / 64;
    for (int it = 0; it < nkb; ++it) {
        if (it + 1 < nkb) issue_kv((it + 1) & 1, (it + 1) * 64);
        const bf16* kl = k_lds[it & 1];
        const bf16* vl = vt_lds[it & 1];

        // S = Q * K^T : 4 key-tiles x (2 k-steps over D)
        v8f s[4];
#pragma unroll
        for (int kt = 0; kt < 4; ++kt) {
            int key = kt * 16 + arow;
            FragAB b0, b1;
            b0.h[0] = *(const v8bf*)&kl[key * 64 + half * 8];
            b0.h[1] = *(const v8bf*)&kl[key * 64 + half * 8 + 16];
            b1.h[0] = *(const v8bf*)&kl[key * 64 + 32 + half * 8];
            b1.h[1] = *(const v8bf*)&kl[key * 64 + 32 + half * 8 + 16];
            v8f z = {};
            z = wmma_bf16(qa[0].v, b0.v, z);
            z = wmma_bf16(qa[1].v, b1.v, z);
            s[kt] = z;
        }

        // Online softmax; emit P (bf16) into per-wave LDS in A-matrix layout.
#pragma unroll
        for (int r = 0; r < 8; ++r) {
            float mx = fmaxf(fmaxf(s[0][r], s[1][r]), fmaxf(s[2][r], s[3][r]));
            mx = rmax16(mx);
            float mnew = fmaxf(mrow[r], mx);
            float corr = __expf(mrow[r] - mnew);
            float p0 = __expf(s[0][r] - mnew);
            float p1 = __expf(s[1][r] - mnew);
            float p2 = __expf(s[2][r] - mnew);
            float p3 = __expf(s[3][r] - mnew);
            float rs = rsum16(p0 + p1 + p2 + p3);
            lrow[r] = lrow[r] * corr + rs;
            mrow[r] = mnew;
#pragma unroll
            for (int dt = 0; dt < 4; ++dt) oacc[dt][r] *= corr;
            int prow = r + half * 8;
            p_lds[w][prow * 64 +  0 + arow] = f2bf(p0);
            p_lds[w][prow * 64 + 16 + arow] = f2bf(p1);
            p_lds[w][prow * 64 + 32 + arow] = f2bf(p2);
            p_lds[w][prow * 64 + 48 + arow] = f2bf(p3);
        }

        // O += P * V : 2 key-chunks x 4 d-tiles (same-wave LDS, in-order)
#pragma unroll
        for (int kc = 0; kc < 2; ++kc) {
            FragAB pa;
            pa.h[0] = *(const v8bf*)&p_lds[w][arow * 64 + kc * 32 + half * 8];
            pa.h[1] = *(const v8bf*)&p_lds[w][arow * 64 + kc * 32 + half * 8 + 16];
#pragma unroll
            for (int dt = 0; dt < 4; ++dt) {
                int drow = dt * 16 + arow;
                FragAB vbf;
                vbf.h[0] = *(const v8bf*)&vl[drow * 64 + kc * 32 + half * 8];
                vbf.h[1] = *(const v8bf*)&vl[drow * 64 + kc * 32 + half * 8 + 16];
                oacc[dt] = wmma_bf16(pa.v, vbf.v, oacc[dt]);
            }
        }

        wait_async0();        // next K/V tiles resident
        __syncthreads();      // all waves done with current buffer
    }

    // Normalize, write bf16 out in [b][n][h*64+d] layout for the proj GEMM.
    const int b = bh >> 4, h = bh & 15;
#pragma unroll
    for (int dt = 0; dt < 4; ++dt)
#pragma unroll
        for (int r = 0; r < 8; ++r) {
            int row = q0 + r + half * 8;
            float val = oacc[dt][r] / lrow[r];
            O[((size_t)b * 2048 + row) * 1024 + h * 64 + dt * 16 + arow] = f2bf(val);
        }
}

// ---------------------------------------------------------------------------
extern "C" void kernel_launch(void* const* d_in, const int* in_sizes, int n_in,
                              void* d_out, int out_size, void* d_ws, size_t ws_size,
                              hipStream_t stream) {
    (void)in_sizes; (void)n_in; (void)out_size; (void)ws_size;
    const float* x     = (const float*)d_in[0];
    const float* Wqkv  = (const float*)d_in[1];
    const float* qg    = (const float*)d_in[2];
    const float* qb    = (const float*)d_in[3];
    const float* kg    = (const float*)d_in[4];
    const float* kb    = (const float*)d_in[5];
    const float* Wproj = (const float*)d_in[6];
    const float* bproj = (const float*)d_in[7];
    float* out = (float*)d_out;

    char* ws = (char*)d_ws;
    bf16*  xb     = (bf16*)ws;  ws += (size_t)4096 * 1024 * 2;
    bf16*  wqkvb  = (bf16*)ws;  ws += (size_t)3072 * 1024 * 2;
    bf16*  wprojb = (bf16*)ws;  ws += (size_t)1024 * 1024 * 2;
    float* qkvf   = (float*)ws; ws += (size_t)4096 * 3072 * 4;
    bf16*  qarr   = (bf16*)ws;  ws += (size_t)32 * 2048 * 64 * 2;
    bf16*  karr   = (bf16*)ws;  ws += (size_t)32 * 2048 * 64 * 2;
    bf16*  vtarr  = (bf16*)ws;  ws += (size_t)32 * 64 * 2048 * 2;
    bf16*  aout   = (bf16*)ws;  ws += (size_t)4096 * 1024 * 2;

    f32_to_bf16_kernel<<<1024, 256, 0, stream>>>(x,     xb,     4096 * 1024);
    f32_to_bf16_kernel<<<1024, 256, 0, stream>>>(Wqkv,  wqkvb,  3072 * 1024);
    f32_to_bf16_kernel<<<512,  256, 0, stream>>>(Wproj, wprojb, 1024 * 1024);

    // qkv = x @ Wqkv^T : [4096,3072]
    gemm_bf16_kernel<<<dim3(64, 24), 256, 0, stream>>>(
        xb, wqkvb, (const float*)nullptr, qkvf, 4096, 3072, 1024);

    ln_pack_kernel<<<8192, 256, 0, stream>>>(qkvf, qg, qb, kg, kb,
                                             qarr, karr, vtarr);

    flash_attn_kernel<<<dim3(32, 32), 128, 0, stream>>>(qarr, karr, vtarr, aout);

    // out = attn_out @ Wproj^T + b : [4096,1024]
    gemm_bf16_kernel<<<dim3(64, 8), 256, 0, stream>>>(
        aout, wprojb, bproj, out, 4096, 1024, 1024);
}
